// TransformerBlock_61538291417735
// MI455X (gfx1250) — matrix operand order repaired and checked by reading, not compiled, not run on hardware
//
#include <hip/hip_runtime.h>

// ---------------------------------------------------------------------------
// Fused Swin-style dual-stream transformer block for MI455X (gfx1250).
// One workgroup = 4 windows, whole block computed in LDS, GEMMs via
// v_wmma_f32_16x16x32_bf16 (bf16 in / f32 accumulate).
// d_ws layout: 2,097,152 bf16 weight elements (4 MB), converted each launch.
// ---------------------------------------------------------------------------

#define Bn     8192
#define Ltok   9
#define Cdim   256
#define NHn    8
#define HDn    32
#define TILEB  4
#define MREAL  (TILEB * Ltok)   // 36 real rows
#define MROWS  48               // padded to 3 WMMA M-tiles
#define MTILES 3
#define WAVES  8
#define SCALEQ 0.17677669529663689f   // 32^-0.5

typedef __attribute__((ext_vector_type(16))) __bf16       v16bf;
typedef __attribute__((ext_vector_type(8)))  float        v8f;
typedef __attribute__((ext_vector_type(4)))  unsigned int u32x4;

union Frag { u32x4 q[2]; v16bf v; };

__device__ __forceinline__ unsigned short f2bf(float f) {
  unsigned int u = __float_as_uint(f);
  u += 0x7FFFu + ((u >> 16) & 1u);           // round-to-nearest-even
  return (unsigned short)(u >> 16);
}
__device__ __forceinline__ float bf2f(unsigned short b) {
  return __uint_as_float(((unsigned int)b) << 16);
}

// A fragment: 16x32 bf16, row = per-lane M row. lane<16: K {k0..k0+7, k0+16..k0+23}
__device__ __forceinline__ v16bf load_frag_a(const unsigned short* base, int ld,
                                             int row, int kt, int lane) {
  const unsigned short* p = base + row * ld + kt * 32 + ((lane & 16) ? 8 : 0);
  Frag f;
  f.q[0] = *(const u32x4*)(p);
  f.q[1] = *(const u32x4*)(p + 16);
  return f.v;
}
// B fragment: 32x16 bf16 from row-major W[N,K]; column n = W row (per-lane).
// lane<16: K {k0..k0+7, k0+16..k0+23}, lane>=16: K {k0+8..k0+15, k0+24..k0+31}.
__device__ __forceinline__ v16bf load_frag_b(const unsigned short* base, int ld,
                                             int nrow, int kt, int lane) {
  const unsigned short* p = base + nrow * ld + kt * 32 + ((lane & 16) ? 8 : 0);
  Frag f;
  f.q[0] = ((const u32x4*)p)[0];
  f.q[1] = *(const u32x4*)(p + 16);
  return f.v;
}
__device__ __forceinline__ v8f wmma_bf16(v16bf a, v16bf b, v8f c) {
  return __builtin_amdgcn_wmma_f32_16x16x32_bf16(false, a, false, b, (short)0, c,
                                                 false, false);
}

struct KP {
  const float *wi, *wb;
  const float *n1g, *n1b, *n2g, *n2b, *n3g, *n3b, *n4g, *n4b;
  const float *qb1, *pb1, *tab1, *qb2, *pb2, *tab2;
  const float *m1b1, *m1b2, *m2b1, *m2b2, *mfb1, *mfb2;
  const float *c1w, *c1b, *c2w, *c2b;
  const unsigned short *Wq1, *Wp1, *Wq2, *Wp2, *W11, *W12, *W21, *W22, *Wf1, *Wf2;
  float* out;
};

// ---------------------------------------------------------------------------

__device__ void load_tile36(const float* g, float* dst) {
  for (int i = threadIdx.x; i < MREAL * Cdim; i += blockDim.x) dst[i] = g[i];
}

// LayerNorm over C=256 of rows in f32 LDS -> bf16 into sA (stride 512).
__device__ void layernorm36(const float* src, const float* gam, const float* bet,
                            unsigned short* dstA) {
  int wave = threadIdx.x >> 5, lane = threadIdx.x & 31;
  for (int r = wave; r < MREAL; r += WAVES) {
    const float* x = src + r * Cdim;
    float s = 0.f;
    for (int c = lane; c < Cdim; c += 32) s += x[c];
#pragma unroll
    for (int o = 16; o > 0; o >>= 1) s += __shfl_xor(s, o);
    float mean = s * (1.0f / Cdim);
    float v = 0.f;
    for (int c = lane; c < Cdim; c += 32) { float d = x[c] - mean; v += d * d; }
#pragma unroll
    for (int o = 16; o > 0; o >>= 1) v += __shfl_xor(v, o);
    float inv = rsqrtf(v * (1.0f / Cdim) + 1e-5f);
    for (int c = lane; c < Cdim; c += 32)
      dstA[r * 512 + c] = f2bf((x[c] - mean) * inv * gam[c] + bet[c]);
  }
}

// QKV (or KV-only) GEMM for one head: N=32 per matrix, K=256, from sA (ld 512).
__device__ void gemm_qkv_head(const unsigned short* sA, const unsigned short* Wq,
                              const float* qb, int h, bool kvOnly, bool saveQ1,
                              float* sqh, float* skh, float* svh,
                              unsigned short* sQ1) {
  int wave = threadIdx.x >> 5, lane = threadIdx.x & 31;
  int jobs = kvOnly ? 12 : 18;
  for (int t = wave; t < jobs; t += WAVES) {
    int which = t / 6 + (kvOnly ? 1 : 0);   // 0=q 1=k 2=v
    int tt = t % 6, mt = tt >> 1, nt = tt & 1;
    int arow = mt * 16 + (lane & 15);
    int feat = which * Cdim + h * HDn + nt * 16 + (lane & 15);
    v8f acc = {};
#pragma unroll
    for (int kt = 0; kt < 8; ++kt)
      acc = wmma_bf16(load_frag_a(sA, 512, arow, kt, lane),
                      load_frag_b(Wq, 256, feat, kt, lane), acc);
    float bias = qb[feat];
    int col = nt * 16 + (lane & 15);
    int radd = (lane & 16) ? 8 : 0;
    float* dst = (which == 0) ? sqh : ((which == 1) ? skh : svh);
#pragma unroll
    for (int r = 0; r < 8; ++r) {
      int row = mt * 16 + radd + r;
      float v = acc[r] + bias;
      if (which == 0) {
        v *= SCALEQ;
        if (saveQ1) sQ1[row * Cdim + h * HDn + col] = f2bf(v);
      }
      dst[row * HDn + col] = v;
    }
  }
}

__device__ void load_qh_from_q1(const unsigned short* sQ1, int h, float* sqh) {
  for (int i = threadIdx.x; i < MREAL * HDn; i += blockDim.x) {
    int r = i >> 5, d = i & 31;
    sqh[i] = bf2f(sQ1[r * Cdim + h * HDn + d]);
  }
}

// Per-head windowed attention (9x9) with Swin relative-position bias; VALU f32.
__device__ void attn_head(const float* tab, int h, const float* sqh,
                          const float* skh, const float* svh, unsigned short* sH) {
  int t = threadIdx.x;
  if (t < MREAL) {
    int win = t / Ltok, i = t % Ltok;
    int rb = win * Ltok;
    float q[HDn];
#pragma unroll
    for (int d = 0; d < HDn; ++d) q[d] = sqh[(rb + i) * HDn + d];
    int ih = i / 3, iw = i % 3;
    float s[Ltok];
    float mx = -3.0e38f;
#pragma unroll
    for (int j = 0; j < Ltok; ++j) {
      float a = 0.f;
      for (int d = 0; d < HDn; ++d) a += q[d] * skh[(rb + j) * HDn + d];
      int rel = (ih - j / 3 + 2) * 5 + (iw - j % 3 + 2);
      a += tab[rel * NHn + h];
      s[j] = a;
      mx = fmaxf(mx, a);
    }
    float sum = 0.f;
#pragma unroll
    for (int j = 0; j < Ltok; ++j) { s[j] = expf(s[j] - mx); sum += s[j]; }
    float inv = 1.f / sum;
#pragma unroll
    for (int d = 0; d < HDn; ++d) {
      float o = 0.f;
      for (int j = 0; j < Ltok; ++j) o += s[j] * svh[(rb + j) * HDn + d];
      sH[(rb + i) * Cdim + h * HDn + d] = f2bf(o * inv);
    }
  }
}

// proj GEMM (N=256,K=256) + bias + residual-add in place into sC (f32).
__device__ void gemm_proj_resid(const unsigned short* sH, const unsigned short* Wp,
                                const float* pb, float* sC) {
  int wave = threadIdx.x >> 5, lane = threadIdx.x & 31;
  for (int t = wave; t < MTILES * 16; t += WAVES) {
    int mt = t >> 4, nt = t & 15;
    int arow = mt * 16 + (lane & 15);
    int wrow = nt * 16 + (lane & 15);
    v8f acc = {};
#pragma unroll
    for (int kt = 0; kt < 8; ++kt)
      acc = wmma_bf16(load_frag_a(sH, 256, arow, kt, lane),
                      load_frag_b(Wp, 256, wrow, kt, lane), acc);
    float bias = pb[wrow];
    int radd = (lane & 16) ? 8 : 0;
    int col = nt * 16 + (lane & 15);
#pragma unroll
    for (int r = 0; r < 8; ++r) sC[(mt * 16 + radd + r) * Cdim + col] += acc[r] + bias;
  }
}

// hidden = GELU(A @ W1_chunk^T + b1_chunk) -> bf16 into sH. N=256.
__device__ void gemm_hidden_gelu(const unsigned short* sA, int lda, int ksteps,
                                 const unsigned short* W1, int ldw, int wrow0,
                                 const float* b1c, unsigned short* sH) {
  int wave = threadIdx.x >> 5, lane = threadIdx.x & 31;
  for (int t = wave; t < MTILES * 16; t += WAVES) {
    int mt = t >> 4, nt = t & 15;
    int arow = mt * 16 + (lane & 15);
    int wrow = wrow0 + nt * 16 + (lane & 15);
    v8f acc = {};
    for (int kt = 0; kt < ksteps; ++kt)
      acc = wmma_bf16(load_frag_a(sA, lda, arow, kt, lane),
                      load_frag_b(W1, ldw, wrow, kt, lane), acc);
    float bias = b1c[nt * 16 + (lane & 15)];
    int radd = (lane & 16) ? 8 : 0;
    int col = nt * 16 + (lane & 15);
#pragma unroll
    for (int r = 0; r < 8; ++r) {
      float x = acc[r] + bias;
      float g = 0.5f * x * (1.f + erff(x * 0.70710678118654752f));  // exact GELU
      sH[(mt * 16 + radd + r) * Cdim + col] = f2bf(g);
    }
  }
}

// Y(f32) += sH @ W2[:, kcol0:kcol0+256]^T.  N=256, K=256.
__device__ void gemm_accum(const unsigned short* sH, const unsigned short* W2,
                           int ldw, int kcol0, float* Y) {
  int wave = threadIdx.x >> 5, lane = threadIdx.x & 31;
  const unsigned short* W2c = W2 + kcol0;
  for (int t = wave; t < MTILES * 16; t += WAVES) {
    int mt = t >> 4, nt = t & 15;
    int arow = mt * 16 + (lane & 15);
    int wrow = nt * 16 + (lane & 15);
    v8f acc = {};
#pragma unroll
    for (int kt = 0; kt < 8; ++kt)
      acc = wmma_bf16(load_frag_a(sH, 256, arow, kt, lane),
                      load_frag_b(W2c, ldw, wrow, kt, lane), acc);
    int radd = (lane & 16) ? 8 : 0;
    int col = nt * 16 + (lane & 15);
#pragma unroll
    for (int r = 0; r < 8; ++r) Y[(mt * 16 + radd + r) * Cdim + col] += acc[r];
  }
}

__device__ void add_bias_from(const float* R, const float* b2, float* Y) {
  for (int i = threadIdx.x; i < MREAL * Cdim; i += blockDim.x) Y[i] = R[i] + b2[i & 255];
}
__device__ void add_bias_inplace(const float* b2, float* Y) {
  for (int i = threadIdx.x; i < MREAL * Cdim; i += blockDim.x) Y[i] += b2[i & 255];
}
__device__ void build_cat(const float* sD, const float* sC, unsigned short* sA) {
  for (int i = threadIdx.x; i < MREAL * Cdim; i += blockDim.x) {
    int r = i >> 8, c = i & 255;
    sA[r * 512 + c]       = f2bf(sD[i]);   // attn_image
    sA[r * 512 + 256 + c] = f2bf(sC[i]);   // image_attn_build
  }
}
// depthwise 3x3 "valid" conv on 3x3 spatial: out[win, c] = sum_t X[t,c]*w[c,t]+b[c]
__device__ void dwconv_out(const float* X, const float* w, const float* b, float* out) {
  for (int i = threadIdx.x; i < TILEB * Cdim; i += blockDim.x) {
    int win = i >> 8, c = i & 255;
    float acc = b[c];
#pragma unroll
    for (int t = 0; t < 9; ++t) acc += X[(win * 9 + t) * Cdim + c] * w[c * 9 + t];
    out[win * Cdim + c] = acc;
  }
}

// ---------------------------------------------------------------------------

__global__ __launch_bounds__(256, 1) void fused_swin_kernel(KP p) {
  __shared__ __align__(16) unsigned short sA[MROWS * 512];   // 48 KB: GEMM A (bf16)
  __shared__ __align__(16) unsigned short sH[MROWS * 256];   // 24 KB: attn-out / hidden chunk
  __shared__ __align__(16) unsigned short sQ1[MROWS * 256];  // 24 KB: saved scaled query
  __shared__ __align__(16) float sC[MROWS * 256];            // 48 KB: residual stream
  __shared__ __align__(16) float sD[MROWS * 256];            // 48 KB: attn_image
  __shared__ __align__(16) float sqh[MROWS * HDn];           // 6 KB per-head q
  __shared__ __align__(16) float skh[MROWS * HDn];           // 6 KB per-head k
  __shared__ __align__(16) float svh[MROWS * HDn];           // 6 KB per-head v

  int wg = blockIdx.x;
  const float* wiT = p.wi + (size_t)wg * MREAL * Cdim;
  const float* wbT = p.wb + (size_t)wg * MREAL * Cdim;

  // ---------------- image stream: _self1 ----------------
  load_tile36(wiT, sC);
  __syncthreads();
  layernorm36(sC, p.n1g, p.n1b, sA);
  __syncthreads();
  for (int h = 0; h < NHn; ++h) {
    gemm_qkv_head(sA, p.Wq1, p.qb1, h, false, true, sqh, skh, svh, sQ1);
    __syncthreads();
    attn_head(p.tab1, h, sqh, skh, svh, sH);
    __syncthreads();
  }
  gemm_proj_resid(sH, p.Wp1, p.pb1, sC);   // sC = x1
  __syncthreads();
  layernorm36(sC, p.n2g, p.n2b, sA);
  add_bias_from(sC, p.m1b2, sD);           // sD = x1 + b2 (Y init)
  __syncthreads();
  for (int kc = 0; kc < 4; ++kc) {         // MLP1 hidden=1024 in 256 chunks
    gemm_hidden_gelu(sA, 512, 8, p.W11, 256, kc * 256, p.m1b1 + kc * 256, sH);
    __syncthreads();
    gemm_accum(sH, p.W12, 1024, kc * 256, sD);
    __syncthreads();
  }
  // sD = attn_image (persists to the end)

  // ---------------- building stream: _self2 ----------------
  load_tile36(wbT, sC);
  __syncthreads();
  layernorm36(sC, p.n3g, p.n3b, sA);
  __syncthreads();
  for (int h = 0; h < NHn; ++h) {
    gemm_qkv_head(sA, p.Wq2, p.qb2, h, false, false, sqh, skh, svh, sQ1);
    __syncthreads();
    attn_head(p.tab2, h, sqh, skh, svh, sH);
    __syncthreads();
  }
  gemm_proj_resid(sH, p.Wp2, p.pb2, sC);   // sC = xb1
  __syncthreads();
  layernorm36(sC, p.n4g, p.n4b, sA);
  __syncthreads();
  add_bias_inplace(p.m2b2, sC);            // in-place Y init
  __syncthreads();
  for (int kc = 0; kc < 2; ++kc) {         // MLP2 hidden=512
    gemm_hidden_gelu(sA, 512, 8, p.W21, 256, kc * 256, p.m2b1 + kc * 256, sH);
    __syncthreads();
    gemm_accum(sH, p.W22, 512, kc * 256, sC);
    __syncthreads();
  }
  // sC = attn_building
  dwconv_out(sC, p.c1w, p.c1b, p.out + (size_t)Bn * Cdim + (size_t)wg * TILEB * Cdim);

  // ---------------- cross: _self2(attn_building, query=q1) ----------------
  layernorm36(sC, p.n3g, p.n3b, sA);
  __syncthreads();
  for (int h = 0; h < NHn; ++h) {
    gemm_qkv_head(sA, p.Wq2, p.qb2, h, true, false, sqh, skh, svh, sQ1);
    load_qh_from_q1(sQ1, h, sqh);          // q comes from saved q1
    __syncthreads();
    attn_head(p.tab2, h, sqh, skh, svh, sH);
    __syncthreads();
  }
  gemm_proj_resid(sH, p.Wp2, p.pb2, sC);   // sC = xc1
  __syncthreads();
  layernorm36(sC, p.n4g, p.n4b, sA);
  __syncthreads();
  add_bias_inplace(p.m2b2, sC);
  __syncthreads();
  for (int kc = 0; kc < 2; ++kc) {
    gemm_hidden_gelu(sA, 512, 8, p.W21, 256, kc * 256, p.m2b1 + kc * 256, sH);
    __syncthreads();
    gemm_accum(sH, p.W22, 512, kc * 256, sC);
    __syncthreads();
  }
  // sC = image_attn_build

  // ---------------- fusion MLP + residual + dwconv ----------------
  build_cat(sD, sC, sA);                   // sA = [attn_image | image_attn_build] bf16
  __syncthreads();
  add_bias_from(sD, p.mfb2, sC);           // Y init = attn_image + mf_b2
  __syncthreads();
  for (int kc = 0; kc < 4; ++kc) {         // mf hidden=1024, K=512
    gemm_hidden_gelu(sA, 512, 16, p.Wf1, 512, kc * 256, p.mfb1 + kc * 256, sH);
    __syncthreads();
    gemm_accum(sH, p.Wf2, 1024, kc * 256, sC);
    __syncthreads();
  }
  // sC = x
  dwconv_out(sC, p.c2w, p.c2b, p.out + (size_t)wg * TILEB * Cdim);
}

// ---------------------------------------------------------------------------
// Weight f32 -> bf16 conversion (prologue, feeds WMMA B operands).
__global__ void convert_f32_to_bf16(const float* __restrict__ src,
                                    unsigned short* __restrict__ dst, int n) {
  int i = blockIdx.x * blockDim.x + threadIdx.x;
  if (i < n) dst[i] = f2bf(src[i]);
}

// ws element offsets (bf16 elems). Total 2,097,152 elems = 4 MB.
static constexpr size_t OFF_Q1 = 0;                     // a1_qkv  [768,256]
static constexpr size_t OFF_P1 = OFF_Q1 + 768 * 256;    // a1_proj [256,256]
static constexpr size_t OFF_Q2 = OFF_P1 + 256 * 256;    // a2_qkv
static constexpr size_t OFF_P2 = OFF_Q2 + 768 * 256;    // a2_proj
static constexpr size_t OFF_11 = OFF_P2 + 256 * 256;    // m1_w1 [1024,256]
static constexpr size_t OFF_12 = OFF_11 + 1024 * 256;   // m1_w2 [256,1024]
static constexpr size_t OFF_21 = OFF_12 + 256 * 1024;   // m2_w1 [512,256]
static constexpr size_t OFF_22 = OFF_21 + 512 * 256;    // m2_w2 [256,512]
static constexpr size_t OFF_F1 = OFF_22 + 256 * 512;    // mf_w1 [1024,512]
static constexpr size_t OFF_F2 = OFF_F1 + 1024 * 512;   // mf_w2 [256,1024]

extern "C" void kernel_launch(void* const* d_in, const int* in_sizes, int n_in,
                              void* d_out, int out_size, void* d_ws, size_t ws_size,
                              hipStream_t stream) {
  (void)in_sizes; (void)n_in; (void)out_size; (void)ws_size;
  unsigned short* wsb = (unsigned short*)d_ws;
  auto cv = [&](int idx, size_t off, int n) {
    convert_f32_to_bf16<<<(n + 255) / 256, 256, 0, stream>>>(
        (const float*)d_in[idx], wsb + off, n);
  };
  cv(10, OFF_Q1, 768 * 256);
  cv(12, OFF_P1, 256 * 256);
  cv(15, OFF_Q2, 768 * 256);
  cv(17, OFF_P2, 256 * 256);
  cv(20, OFF_11, 1024 * 256);
  cv(22, OFF_12, 256 * 1024);
  cv(24, OFF_21, 512 * 256);
  cv(26, OFF_22, 256 * 512);
  cv(28, OFF_F1, 1024 * 512);
  cv(30, OFF_F2, 256 * 1024);

  KP p;
  p.wi  = (const float*)d_in[0];  p.wb  = (const float*)d_in[1];
  p.n1g = (const float*)d_in[2];  p.n1b = (const float*)d_in[3];
  p.n2g = (const float*)d_in[4];  p.n2b = (const float*)d_in[5];
  p.n3g = (const float*)d_in[6];  p.n3b = (const float*)d_in[7];
  p.n4g = (const float*)d_in[8];  p.n4b = (const float*)d_in[9];
  p.qb1 = (const float*)d_in[11]; p.pb1 = (const float*)d_in[13];
  p.tab1 = (const float*)d_in[14];
  p.qb2 = (const float*)d_in[16]; p.pb2 = (const float*)d_in[18];
  p.tab2 = (const float*)d_in[19];
  p.m1b1 = (const float*)d_in[21]; p.m1b2 = (const float*)d_in[23];
  p.m2b1 = (const float*)d_in[25]; p.m2b2 = (const float*)d_in[27];
  p.mfb1 = (const float*)d_in[29]; p.mfb2 = (const float*)d_in[31];
  p.c1w = (const float*)d_in[32]; p.c1b = (const float*)d_in[33];
  p.c2w = (const float*)d_in[34]; p.c2b = (const float*)d_in[35];
  p.Wq1 = wsb + OFF_Q1; p.Wp1 = wsb + OFF_P1;
  p.Wq2 = wsb + OFF_Q2; p.Wp2 = wsb + OFF_P2;
  p.W11 = wsb + OFF_11; p.W12 = wsb + OFF_12;
  p.W21 = wsb + OFF_21; p.W22 = wsb + OFF_22;
  p.Wf1 = wsb + OFF_F1; p.Wf2 = wsb + OFF_F2;
  p.out = (float*)d_out;

  fused_swin_kernel<<<Bn / TILEB, 256, 0, stream>>>(p);
}
